// CrossLayerAttention_24292335026337
// MI455X (gfx1250) — matrix operand order repaired and checked by reading, not compiled, not run on hardware
//
#include <hip/hip_runtime.h>

// CDNA5 / gfx1250, wave32. Heavy contractions on v_wmma_f32_16x16x32_bf16;
// GEMM B-operand staged via async global->LDS copies (ASYNCcnt) when available.

typedef __bf16 bf16;
typedef __attribute__((ext_vector_type(16))) __bf16 v16bf;
typedef __attribute__((ext_vector_type(8)))  __bf16 v8bf;
typedef __attribute__((ext_vector_type(8)))  float  v8f;
typedef __attribute__((ext_vector_type(4)))  int    v4i;

#define L_  2048
#define C_  2048
#define H_  16
#define D_  128
#define KL_ 4096            // 2*L keys after concat
#define LC_ (L_*C_)

#if defined(__gfx1250__) && __has_builtin(__builtin_amdgcn_global_load_async_to_lds_b128)
#define USE_ASYNC_B 1
// builtin wants (int4 addrspace(1)*, int4 addrspace(3)*, imm offset, imm cpol)
#define ASYNC_B128(g, l)                                                   \
  __builtin_amdgcn_global_load_async_to_lds_b128(                          \
      (__attribute__((address_space(1))) v4i*)(g),                         \
      (__attribute__((address_space(3))) v4i*)(l), 0, 0)
#else
#define USE_ASYNC_B 0
#endif

#if __has_builtin(__builtin_amdgcn_s_wait_asynccnt)
#define WAIT_ASYNC(n) __builtin_amdgcn_s_wait_asynccnt(n)
#else
#define WAIT_ASYNC(n) asm volatile("s_wait_asynccnt " #n ::: "memory")
#endif

union AFrag { v16bf v; v8bf half8[2]; };

__device__ __forceinline__ v8f wmma_bf16(v16bf a, v16bf b, v8f c) {
  // (neg_a, A, neg_b, B, c_mod, C, reuse_a, reuse_b)
  return __builtin_amdgcn_wmma_f32_16x16x32_bf16(false, a, false, b, (short)0, c,
                                                 false, false);
}

// ---------------- elementwise converts ----------------
__global__ void k_cvt(const float* __restrict__ in, bf16* __restrict__ out, int n) {
  int i = blockIdx.x * blockDim.x + threadIdx.x;
  if (i < n) out[i] = (bf16)in[i];
}

// sw2 is (C,C) contracted over its FIRST index; store transposed so the GEMM's
// B operand has the contraction contiguous, same as Wq/Wk/Wv/Wo.
__global__ void k_cvt_t(const float* __restrict__ in, bf16* __restrict__ out) {
  int i = blockIdx.x * blockDim.x + threadIdx.x;       // over C*C
  int r = i >> 11, c = i & (C_ - 1);
  out[i] = (bf16)in[c * C_ + r];                       // out[r][c] = in[c][r]
}

// pos activation: s[l,c] = sin(W0*(coord[l]*sw1[c] + sb1[c]))
__global__ void k_sinpos(const float* __restrict__ sw1, const float* __restrict__ sb1,
                         bf16* __restrict__ out) {
  int i = blockIdx.x * blockDim.x + threadIdx.x;       // over L*C
  int l = i >> 11, c = i & (C_ - 1);
  float coord = (float)l * (2.0f / (float)(L_ - 1)) - 1.0f;
  out[i] = (bf16)__sinf(30.0f * (coord * sw1[c] + sb1[c]));
}

// ---------------- bf16 WMMA GEMM:  C[M,N] (f32) = A[M,K] * B[N,K]^T ----------------
// 128x128 block tile, 8 waves (4x2), each wave 32x64 = 2x4 wmma tiles.
// B tile (128 rows x 32 k = 8KB) staged to LDS with double-buffered async copies
// (steady-state loop is branch-free: always issue next tile, last tile peeled).
// A fragments loaded directly (only 2x reuse, pipelines well from global).
__global__ __launch_bounds__(256) void k_gemm(const bf16* __restrict__ A,
                                              const bf16* __restrict__ B,
                                              float* __restrict__ Cd,
                                              int K, int lda, int ldb, int ldc,
                                              int accum) {
  int tid  = threadIdx.x;
  int lane = tid & 31;
  int wave = tid >> 5;
  int wm = wave >> 1, wn = wave & 1;
  int lo = lane & 15, hi = lane >> 4;
  int rowBase  = blockIdx.y * 128 + wm * 32;
  int colBlock = blockIdx.x * 128;
  int colBase  = colBlock + wn * 64;

  v8f acc[2][4];
  #pragma unroll
  for (int a = 0; a < 2; a++)
    #pragma unroll
    for (int b = 0; b < 4; b++) acc[a][b] = (v8f){0,0,0,0,0,0,0,0};

#if USE_ASYNC_B
  __shared__ __align__(16) bf16 Bs[2][128 * 32];
  // copy role: each of 256 threads moves 32B of the 8KB tile (2x b128)
  int crow = tid >> 1;               // 0..127 (feature row within block)
  int cseg = (tid & 1) * 16;         // element offset 0 / 16 within 32-k chunk
  const bf16* gB = B + (size_t)(colBlock + crow) * ldb + cseg;
  bf16* lB0 = &Bs[0][crow * 32 + cseg];
  bf16* lB1 = &Bs[1][crow * 32 + cseg];

  auto compute_step = [&](int kk, const bf16* lbuf) {
    AFrag af[2];
    #pragma unroll
    for (int tm = 0; tm < 2; tm++) {
      const bf16* ap = A + (size_t)(rowBase + tm * 16 + lo) * lda + kk + hi * 16;
      af[tm].half8[0] = *(const v8bf*)(ap);
      af[tm].half8[1] = *(const v8bf*)(ap + 8);
    }
    #pragma unroll
    for (int tn = 0; tn < 4; tn++) {
      v16bf bfrag = *(const v16bf*)&lbuf[(wn * 64 + tn * 16 + lo) * 32 + hi * 16];
      #pragma unroll
      for (int tm = 0; tm < 2; tm++)
        acc[tm][tn] = wmma_bf16(af[tm].v, bfrag, acc[tm][tn]);
    }
  };

  // prologue: tile 0 into buffer 0
  ASYNC_B128(gB,     lB0);
  ASYNC_B128(gB + 8, lB0 + 8);

  int buf = 0;
  int k0 = 0;
  // steady state: always one tile in flight, no branches in the body
  for (; k0 < K - 32; k0 += 32) {
    const bf16* gn = gB + k0 + 32;
    bf16* ln = buf ? lB0 : lB1;
    ASYNC_B128(gn,     ln);
    ASYNC_B128(gn + 8, ln + 8);
    WAIT_ASYNC(2);                  // current tile landed; next still in flight
    __syncthreads();
    compute_step(k0, buf ? &Bs[1][0] : &Bs[0][0]);
    __syncthreads();
    buf ^= 1;
  }
  // epilogue: last tile
  WAIT_ASYNC(0);
  __syncthreads();
  compute_step(k0, buf ? &Bs[1][0] : &Bs[0][0]);
#else
  for (int k0 = 0; k0 < K; k0 += 32) {
    AFrag af[2];
    #pragma unroll
    for (int tm = 0; tm < 2; tm++) {
      const bf16* ap = A + (size_t)(rowBase + tm * 16 + lo) * lda + k0 + hi * 16;
      af[tm].half8[0] = *(const v8bf*)(ap);
      af[tm].half8[1] = *(const v8bf*)(ap + 8);
    }
    #pragma unroll
    for (int tn = 0; tn < 4; tn++) {
      const bf16* bp = B + (size_t)(colBase + tn * 16 + lo) * ldb + k0 + hi * 16;
      v16bf bfrag = *(const v16bf*)bp;
      #pragma unroll
      for (int tm = 0; tm < 2; tm++)
        acc[tm][tn] = wmma_bf16(af[tm].v, bfrag, acc[tm][tn]);
    }
  }
#endif

  #pragma unroll
  for (int tm = 0; tm < 2; tm++)
    #pragma unroll
    for (int tn = 0; tn < 4; tn++)
      #pragma unroll
      for (int r = 0; r < 8; r++) {
        int row = rowBase + tm * 16 + r + hi * 8;
        int col = colBase + tn * 16 + lo;
        float v = acc[tm][tn][r];
        if (accum) v += Cd[(size_t)row * ldc + col];
        Cd[(size_t)row * ldc + col] = v;
      }
}

// ---------------- per-(l,h) rmsnorm over D + depth-RoPE, f32 -> bf16 ----------------
// One wave per (l,h). Lane owns d = {lane, lane+32, lane+64, lane+96}.
__global__ void k_prep_qk(const float* __restrict__ in, bf16* __restrict__ out,
                          const float* __restrict__ w, float depth, float outScale,
                          int rowOff) {
  int l = blockIdx.x / H_, h = blockIdx.x % H_;
  int lane = threadIdx.x;
  const float* p = in + (size_t)l * C_ + h * D_;
  float x0 = p[lane], x1 = p[lane + 32], x2 = p[lane + 64], x3 = p[lane + 96];
  float ss = x0 * x0 + x1 * x1 + x2 * x2 + x3 * x3;
  #pragma unroll
  for (int m = 1; m < 32; m <<= 1) ss += __shfl_xor(ss, m, 32);
  float rs = rsqrtf(ss * (1.0f / (float)D_) + 1e-6f);
  float n0 = x0 * rs * w[lane],      n1 = x1 * rs * w[lane + 32];
  float n2 = x2 * rs * w[lane + 64], n3 = x3 * rs * w[lane + 96];
  // freqs_j = 10000^(-j/64); pairs (j, j+64)
  const float LN1E4_64 = 9.210340371976184f / 64.0f;
  float a0 = depth * __expf(-(float)lane * LN1E4_64);
  float a1 = depth * __expf(-(float)(lane + 32) * LN1E4_64);
  float c0 = __cosf(a0), s0 = __sinf(a0), c1 = __cosf(a1), s1 = __sinf(a1);
  bf16* q = out + (size_t)(rowOff + l) * C_ + h * D_;
  q[lane]      = (bf16)((n0 * c0 - n2 * s0) * outScale);
  q[lane + 32] = (bf16)((n1 * c1 - n3 * s1) * outScale);
  q[lane + 64] = (bf16)((n0 * s0 + n2 * c0) * outScale);
  q[lane + 96] = (bf16)((n1 * s1 + n3 * c1) * outScale);
}

// V transpose: vt[(h*D+d)*KL + key] = v[l][h*D+d]   (key contiguous for PV B-fragment)
__global__ void k_prep_v(const float* __restrict__ in, bf16* __restrict__ vt, int rowOff) {
  int i = blockIdx.x * blockDim.x + threadIdx.x;       // over L*C
  int l = i >> 11, c = i & (C_ - 1);
  vt[(size_t)c * KL_ + rowOff + l] = (bf16)in[i];
}

// ---------------- flash attention: per-head, 16 q rows per wave ----------------
__global__ __launch_bounds__(128) void k_flash(const bf16* __restrict__ qb,
                                               const bf16* __restrict__ kb,
                                               const bf16* __restrict__ vt,
                                               bf16* __restrict__ ob) {
  __shared__ __align__(16) bf16 ldsP[4][16 * 32];      // per-wave P staging (16x32)
  int lane = threadIdx.x & 31, wave = threadIdx.x >> 5;
  int lo = lane & 15, hi = lane >> 4;
  int h = blockIdx.y;
  int q0 = (blockIdx.x * 4 + wave) * 16;

  // preload Q fragments (16 rows x 128 dim -> 4 K-chunks of 32); 1/sqrt(D) folded in.
  AFrag aq[4];
  #pragma unroll
  for (int c = 0; c < 4; c++) {
    const bf16* p = qb + (size_t)(q0 + lo) * C_ + h * D_ + c * 32 + hi * 16;
    aq[c].half8[0] = *(const v8bf*)p;
    aq[c].half8[1] = *(const v8bf*)(p + 8);
  }

  float mrun[8], lrun[8];
  v8f acco[8];
  #pragma unroll
  for (int r = 0; r < 8; r++) { mrun[r] = -1e30f; lrun[r] = 0.0f; }
  #pragma unroll
  for (int nc = 0; nc < 8; nc++) acco[nc] = (v8f){0,0,0,0,0,0,0,0};

  for (int kb0 = 0; kb0 < KL_; kb0 += 32) {
    // S = Q * K^T for 32 keys (two 16x16 tiles), contraction over D=128
    v8f s0 = (v8f){0,0,0,0,0,0,0,0}, s1 = (v8f){0,0,0,0,0,0,0,0};
    #pragma unroll
    for (int c = 0; c < 4; c++) {
      const bf16* kp = kb + (size_t)(kb0 + lo) * C_ + h * D_ + c * 32 + hi * 16;
      v16bf b0 = *(const v16bf*)kp;
      v16bf b1 = *(const v16bf*)(kp + (size_t)16 * C_);
      s0 = wmma_bf16(aq[c].v, b0, s0);
      s1 = wmma_bf16(aq[c].v, b1, s1);
    }
    // online softmax; rows split VGPR-wise (r / r+8), cols across 16-lane halves
    #pragma unroll
    for (int r = 0; r < 8; r++) {
      float mx = fmaxf(s0[r], s1[r]);
      #pragma unroll
      for (int m = 1; m < 16; m <<= 1) mx = fmaxf(mx, __shfl_xor(mx, m, 32));
      float mnew = fmaxf(mrun[r], mx);
      float sc = __expf(mrun[r] - mnew);
      mrun[r] = mnew;
      float e0 = __expf(s0[r] - mnew), e1 = __expf(s1[r] - mnew);
      float rsum = e0 + e1;
      #pragma unroll
      for (int m = 1; m < 16; m <<= 1) rsum += __shfl_xor(rsum, m, 32);
      lrun[r] = lrun[r] * sc + rsum;
      #pragma unroll
      for (int nc = 0; nc < 8; nc++) acco[nc][r] *= sc;
      int row = r + hi * 8;
      ldsP[wave][row * 32 + lo]      = (bf16)e0;
      ldsP[wave][row * 32 + 16 + lo] = (bf16)e1;
    }
    __syncthreads();
    // re-layout P (C/D layout) -> A-fragment layout via LDS
    AFrag ap;
    ap.half8[0] = *(const v8bf*)&ldsP[wave][lo * 32 + hi * 16];
    ap.half8[1] = *(const v8bf*)&ldsP[wave][lo * 32 + hi * 16 + 8];
    // O += P * V  (V pre-transposed: keys contiguous per (h,d) row)
    #pragma unroll
    for (int nc = 0; nc < 8; nc++) {
      const bf16* vp = vt + (size_t)(h * D_ + nc * 16 + lo) * KL_ + kb0 + hi * 16;
      v16bf bv = *(const v16bf*)vp;
      acco[nc] = wmma_bf16(ap.v, bv, acco[nc]);
    }
    __syncthreads();
  }
  #pragma unroll
  for (int r = 0; r < 8; r++) {
    float inv = 1.0f / lrun[r];
    int row = q0 + r + hi * 8;
    #pragma unroll
    for (int nc = 0; nc < 8; nc++)
      ob[(size_t)row * C_ + h * D_ + nc * 16 + lo] = (bf16)(acco[nc][r] * inv);
  }
}

// ---------------- final: out = x + rmsnorm(t + sb2, on_w) ----------------
__global__ __launch_bounds__(256) void k_final(const float* __restrict__ t,
                                               const float* __restrict__ sb2,
                                               const float* __restrict__ onw,
                                               const float* __restrict__ x,
                                               float* __restrict__ out) {
  __shared__ float red[8];
  int l = blockIdx.x;
  int tid = threadIdx.x, lane = tid & 31, wave = tid >> 5;
  const float* tr = t + (size_t)l * C_;
  float ss = 0.0f;
  #pragma unroll
  for (int i = 0; i < 8; i++) {
    int c = tid + i * 256;
    float v = tr[c] + sb2[c];
    ss += v * v;
  }
  #pragma unroll
  for (int m = 1; m < 32; m <<= 1) ss += __shfl_xor(ss, m, 32);
  if (lane == 0) red[wave] = ss;
  __syncthreads();
  float tot = 0.0f;
  #pragma unroll
  for (int w = 0; w < 8; w++) tot += red[w];
  float rs = rsqrtf(tot * (1.0f / (float)C_) + 1e-6f);
  #pragma unroll
  for (int i = 0; i < 8; i++) {
    int c = tid + i * 256;
    float v = tr[c] + sb2[c];
    out[(size_t)l * C_ + c] = x[(size_t)l * C_ + c] + v * rs * onw[c];
  }
}

// ---------------- host orchestration ----------------
extern "C" void kernel_launch(void* const* d_in, const int* in_sizes, int n_in,
                              void* d_out, int out_size, void* d_ws, size_t ws_size,
                              hipStream_t stream) {
  const float* x   = (const float*)d_in[0];
  const float* h0  = (const float*)d_in[1];
  const float* h1  = (const float*)d_in[2];
  const float* h2  = (const float*)d_in[3];
  const float* Wq  = (const float*)d_in[4];
  const float* Wk  = (const float*)d_in[5];
  const float* Wv  = (const float*)d_in[6];
  const float* Wo  = (const float*)d_in[7];
  const float* qn  = (const float*)d_in[8];
  const float* kn  = (const float*)d_in[9];
  const float* onw = (const float*)d_in[10];
  const float* sw1 = (const float*)d_in[11];
  const float* sb1 = (const float*)d_in[12];
  const float* sw2 = (const float*)d_in[13];
  const float* sb2 = (const float*)d_in[14];
  float* out = (float*)d_out;

  char* base = (char*)d_ws;
  size_t off = 0;
  auto take = [&](size_t bytes) -> char* {
    char* p = base + off;
    off += (bytes + 255) & ~(size_t)255;
    return p;
  };
  bf16*  h0b  = (bf16*)take((size_t)LC_ * 2);
  bf16*  h1b  = (bf16*)take((size_t)LC_ * 2);
  bf16*  h2b  = (bf16*)take((size_t)LC_ * 2);
  bf16*  Wqb  = (bf16*)take((size_t)LC_ * 2);
  bf16*  Wkb  = (bf16*)take((size_t)LC_ * 2);
  bf16*  Wvb  = (bf16*)take((size_t)LC_ * 2);
  bf16*  Wob  = (bf16*)take((size_t)LC_ * 2);
  bf16*  sw2b = (bf16*)take((size_t)LC_ * 2);     // transposed
  bf16*  qb   = (bf16*)take((size_t)LC_ * 2);
  bf16*  kb   = (bf16*)take((size_t)2 * LC_ * 2);
  bf16*  vt   = (bf16*)take((size_t)2 * LC_ * 2);
  bf16*  aob  = (bf16*)take((size_t)LC_ * 2);     // attention out (bf16 for O-proj)
  bf16*  sact = (bf16*)take((size_t)LC_ * 2);     // sin activation
  float* gbuf = (float*)take((size_t)LC_ * 4);    // reused f32 GEMM out + final t

  const int EW = LC_ / 256;                       // elementwise grid
  dim3 gg(C_ / 128, L_ / 128);                    // GEMM grid (N tiles, M tiles)

  // bf16 conversions
  k_cvt  <<<EW, 256, 0, stream>>>(h0, h0b, LC_);
  k_cvt  <<<EW, 256, 0, stream>>>(h1, h1b, LC_);
  k_cvt  <<<EW, 256, 0, stream>>>(h2, h2b, LC_);
  k_cvt  <<<EW, 256, 0, stream>>>(Wq, Wqb, LC_);
  k_cvt  <<<EW, 256, 0, stream>>>(Wk, Wkb, LC_);
  k_cvt  <<<EW, 256, 0, stream>>>(Wv, Wvb, LC_);
  k_cvt  <<<EW, 256, 0, stream>>>(Wo, Wob, LC_);
  k_cvt_t<<<EW, 256, 0, stream>>>(sw2, sw2b);

  const float QSCALE = 0.08838834764831845f;      // 1/sqrt(D) folded into q

  // Q = h2 @ Wq^T -> rmsnorm+rope(depth=2), scaled
  k_gemm<<<gg, 256, 0, stream>>>(h2b, Wqb, gbuf, C_, C_, C_, C_, 0);
  k_prep_qk<<<L_ * H_, 32, 0, stream>>>(gbuf, qb, qn, 2.0f, QSCALE, 0);
  // K0 = h0 @ Wk^T (depth 0), K1 = h1 @ Wk^T (depth 1), concat rows
  k_gemm<<<gg, 256, 0, stream>>>(h0b, Wkb, gbuf, C_, C_, C_, C_, 0);
  k_prep_qk<<<L_ * H_, 32, 0, stream>>>(gbuf, kb, kn, 0.0f, 1.0f, 0);
  k_gemm<<<gg, 256, 0, stream>>>(h1b, Wkb, gbuf, C_, C_, C_, C_, 0);
  k_prep_qk<<<L_ * H_, 32, 0, stream>>>(gbuf, kb, kn, 1.0f, 1.0f, L_);
  // V0, V1 -> transposed bf16
  k_gemm<<<gg, 256, 0, stream>>>(h0b, Wvb, gbuf, C_, C_, C_, C_, 0);
  k_prep_v<<<EW, 256, 0, stream>>>(gbuf, vt, 0);
  k_gemm<<<gg, 256, 0, stream>>>(h1b, Wvb, gbuf, C_, C_, C_, C_, 0);
  k_prep_v<<<EW, 256, 0, stream>>>(gbuf, vt, L_);

  // fused flash attention over 4096 keys
  k_flash<<<dim3(L_ / 64, H_), 128, 0, stream>>>(qb, kb, vt, aob);

  // pos activation, then: gbuf = attn_out @ Wo^T ; gbuf += sin_act @ sw2
  k_sinpos<<<EW, 256, 0, stream>>>(sw1, sb1, sact);
  k_gemm<<<gg, 256, 0, stream>>>(aob,  Wob,  gbuf, C_, C_, C_, C_, 0);
  k_gemm<<<gg, 256, 0, stream>>>(sact, sw2b, gbuf, C_, C_, C_, C_, 1);

  // out = x + rmsnorm(gbuf + sb2, on_w)
  k_final<<<L_, 256, 0, stream>>>(gbuf, sb2, onw, x, out);
}
